// ValueMoE_37391985279698
// MI455X (gfx1250) — matrix operand order repaired and checked by reading, not compiled, not run on hardware
//
#include <hip/hip_runtime.h>
#include <hip/hip_bf16.h>

// ---------------------------------------------------------------------------
// ValueMoE top-1 routed ternary-masked GEMM for MI455X (gfx1250, wave32).
// T=4096 tokens; K=2048; N=2048; E=32; THRESH=0.5.
// Per-expert gathered GEMM, 128x128 output tile per block, bf16 WMMA,
// double-buffered global_load_async_to_lds_b128 pipeline (ASYNCcnt).
// ---------------------------------------------------------------------------

#define T_TOKENS 4096
#define K_DIM    2048
#define N_DIM    2048
#define N_EXP    32

#define TILE_M 128
#define TILE_N 128
#define TILE_K 32

// LDS layout (bytes):
//   raw[buf] : A fp32 16384 | W fp32 16384 | M fp32 16384   (buf stride 49152)
//   raw[0] @ 0, raw[1] @ 49152
//   lA bf16 [128][32] @ 98304  (8192 B)
//   lB bf16 [32][128] @ 106496 (8192 B)
//   toks     @ 114688 (512 B), tw @ 115200 (512 B)
#define RAW_STRIDE 49152
#define RAW_W_OFF  16384
#define RAW_M_OFF  32768
#define LA_OFF     98304
#define LB_OFF     106496
#define TOK_OFF    114688
#define TW_OFF     115200
#define SMEM_BYTES 115712

typedef __attribute__((ext_vector_type(16))) __bf16 v16bf;
typedef __attribute__((ext_vector_type(8)))  float  v8f;

typedef __attribute__((address_space(3))) const char lds_char_t;

__device__ static inline uint32_t to_lds(const void* p) {
    return (uint32_t)(uintptr_t)(lds_char_t*)p;
}

// Async copy 16B/lane global->LDS, GV mode (per-lane 64-bit address).
__device__ static inline void async_b128_gv(uint32_t lds_addr, uint64_t gaddr) {
    asm volatile("global_load_async_to_lds_b128 %0, %1, off"
                 :: "v"(lds_addr), "v"(gaddr) : "memory");
}
// Async copy 16B/lane global->LDS, GVS mode (SGPR base + 32-bit VGPR offset).
__device__ static inline void async_b128_gvs(uint32_t lds_addr, uint32_t voff,
                                             uint64_t sbase) {
    asm volatile("global_load_async_to_lds_b128 %0, %1, %2"
                 :: "v"(lds_addr), "v"(voff), "s"(sbase) : "memory");
}
__device__ static inline void wait_async0() {
    asm volatile("s_wait_asynccnt 0x0" ::: "memory");
}

// ---------------------------------------------------------------------------
// Kernel 0: zero the per-expert counters.
// ---------------------------------------------------------------------------
__global__ void moe_init(int* __restrict__ counts) {
    if (threadIdx.x < N_EXP) counts[threadIdx.x] = 0;
}

// ---------------------------------------------------------------------------
// Kernel 1: router.  One wave per token, lane == expert (E == 32 == wave32).
// ---------------------------------------------------------------------------
__global__ __launch_bounds__(256) void moe_router(
    const float* __restrict__ x,
    const float* __restrict__ router_w,
    float* __restrict__ top1w,
    int*   __restrict__ top1i,
    int*   __restrict__ counts)
{
    const int wave = threadIdx.x >> 5;
    const int lane = threadIdx.x & 31;
    const int t    = blockIdx.x * 8 + wave;
    if (t >= T_TOKENS) return;

    const float* xr = x + (size_t)t * K_DIM;
    const float* rr = router_w + (size_t)lane * K_DIM;

    float acc = 0.0f;
    for (int k = 0; k < K_DIM; k += 4) {
        float4 xv = *(const float4*)(xr + k);
        float4 rv = *(const float4*)(rr + k);
        acc += xv.x * rv.x + xv.y * rv.y + xv.z * rv.z + xv.w * rv.w;
    }

    float vmax = acc;
    int   imax = lane;
    #pragma unroll
    for (int off = 16; off > 0; off >>= 1) {
        float ov = __shfl_xor(vmax, off, 32);
        int   oi = __shfl_xor(imax, off, 32);
        if (ov > vmax || (ov == vmax && oi < imax)) { vmax = ov; imax = oi; }
    }

    float s = __expf(acc - vmax);
    #pragma unroll
    for (int off = 16; off > 0; off >>= 1) s += __shfl_xor(s, off, 32);

    if (lane == 0) {
        top1w[t] = 1.0f / s;
        top1i[t] = imax;
        atomicAdd(&counts[imax], 1);
    }
}

// ---------------------------------------------------------------------------
// Kernel 2: exclusive scan over 32 expert counts.
// ---------------------------------------------------------------------------
__global__ void moe_scan(const int* __restrict__ counts,
                         int* __restrict__ seg, int* __restrict__ cur) {
    if (threadIdx.x == 0) {
        int acc = 0;
        for (int e = 0; e < N_EXP; ++e) {
            seg[e] = acc;
            cur[e] = acc;
            acc += counts[e];
        }
    }
}

// ---------------------------------------------------------------------------
// Kernel 3: scatter token ids into per-expert contiguous segments.
// ---------------------------------------------------------------------------
__global__ __launch_bounds__(256) void moe_scatter(
    const int* __restrict__ top1i, int* __restrict__ cur,
    int* __restrict__ tlist) {
    int t = blockIdx.x * 256 + threadIdx.x;
    if (t >= T_TOKENS) return;
    int e   = top1i[t];
    int pos = atomicAdd(&cur[e], 1);
    tlist[pos] = t;
}

// ---------------------------------------------------------------------------
// Kernel 4: per-expert gathered GEMM with bf16 WMMA + async-to-LDS pipeline.
//   grid = (N_DIM/TILE_N, T_TOKENS/TILE_M, N_EXP); block = 256 (8 waves).
//   Wave w owns N cols [w*16, w*16+16) x all 128 M rows: 8 WMMA acc tiles,
//   B operand reused across the 8 A sub-tiles each K-step.
// ---------------------------------------------------------------------------
__global__ __launch_bounds__(256) void moe_gemm(
    const float* __restrict__ x,       // [T][K]
    const float* __restrict__ weight,  // [N][K]
    const float* __restrict__ scale,   // [N]
    const float* __restrict__ masks,   // [E][N][K]
    const float* __restrict__ top1w,   // [T]
    const int*   __restrict__ seg,     // [E]
    const int*   __restrict__ counts,  // [E]
    const int*   __restrict__ tlist,   // [T]
    float*       __restrict__ out)     // [T][N]
{
    const int e   = blockIdx.z;
    const int cnt = counts[e];
    const int m0  = blockIdx.y * TILE_M;
    if (m0 >= cnt) return;
    const int nbase = blockIdx.x * TILE_N;
    const int sbase = seg[e];

    __shared__ __align__(16) char smem[SMEM_BYTES];
    int*    toks = (int*)(smem + TOK_OFF);
    float*  twp  = (float*)(smem + TW_OFF);
    __bf16* lA   = (__bf16*)(smem + LA_OFF);   // [128][32]
    __bf16* lB   = (__bf16*)(smem + LB_OFF);   // [32][128]

    const int tid  = threadIdx.x;
    const int lane = tid & 31;
    const int wave = tid >> 5;

    if (tid < TILE_M) {
        int m = m0 + tid;
        int t = (m < cnt) ? tlist[sbase + m] : tlist[sbase];  // clamp pad rows
        toks[tid] = t;
        twp[tid]  = top1w[t];
    }
    __syncthreads();

    // ---- per-wave async-issue addressing ----------------------------------
    // Each wave copies rows [wave*16, wave*16+16) of the A/W/M 128x32 tiles.
    // One b128 instruction moves 4 rows (8 lanes x 16B per row).
    const int sub  = lane >> 3;          // 0..3: row within the 4-row group
    const int colb = (lane & 7) * 16;    // byte offset within a 128B row
    const uint32_t lds_base = to_lds(smem);

    uint64_t aA[4];                      // per-lane 64-bit x addresses (GV)
    uint32_t vWM[4];                     // per-lane 32-bit W/M offsets (GVS)
    uint32_t ldsA[4], ldsW[4], ldsM[4];  // LDS byte addresses (buf 0)
    #pragma unroll
    for (int j = 0; j < 4; ++j) {
        int m = wave * 16 + j * 4 + sub;             // 0..127
        aA[j]  = (uint64_t)(uintptr_t)x + (uint64_t)toks[m] * (K_DIM * 4) + colb;
        vWM[j] = (uint32_t)(m * (K_DIM * 4) + colb);
        ldsA[j] = lds_base + (uint32_t)(m * 128 + colb);
        ldsW[j] = ldsA[j] + RAW_W_OFF;
        ldsM[j] = ldsA[j] + RAW_M_OFF;
    }
    const uint64_t sW = (uint64_t)(uintptr_t)weight + (uint64_t)nbase * (K_DIM * 4);
    const uint64_t sM = (uint64_t)(uintptr_t)masks +
                        ((uint64_t)e * N_DIM + nbase) * (K_DIM * 4);

    // ---- conversion-phase mapping -----------------------------------------
    const int cm  = tid >> 1;            // 0..127 (row for both A and W/M)
    const int ckh = (tid & 1) * 16;      // 0 or 16 (k half)
    const float sc = scale[nbase + cm];

    // ---- WMMA lane mapping -------------------------------------------------
    const int hi = (lane >= 16) ? 1 : 0;
    const int ar = lane & 15;

    v8f c[8];
    #pragma unroll
    for (int mt = 0; mt < 8; ++mt) c[mt] = (v8f){};

    // prologue: kick off tile 0 into raw buffer 0
    #pragma unroll
    for (int j = 0; j < 4; ++j) async_b128_gv (ldsA[j], aA[j]);
    #pragma unroll
    for (int j = 0; j < 4; ++j) async_b128_gvs(ldsW[j], vWM[j], sW);
    #pragma unroll
    for (int j = 0; j < 4; ++j) async_b128_gvs(ldsM[j], vWM[j], sM);
    #pragma unroll
    for (int j = 0; j < 4; ++j) { aA[j] += TILE_K * 4; vWM[j] += TILE_K * 4; }

    uint32_t bufoff = 0;
    for (int k0 = 0; k0 < K_DIM; k0 += TILE_K) {
        wait_async0();           // this wave's raw[buf] slice landed in LDS
        __syncthreads();         // all slices landed; prev wmma reads done

        if (k0 + TILE_K < K_DIM) {
            uint32_t nb = bufoff ^ RAW_STRIDE;
            #pragma unroll
            for (int j = 0; j < 4; ++j) async_b128_gv (ldsA[j] + nb, aA[j]);
            #pragma unroll
            for (int j = 0; j < 4; ++j) async_b128_gvs(ldsW[j] + nb, vWM[j], sW);
            #pragma unroll
            for (int j = 0; j < 4; ++j) async_b128_gvs(ldsM[j] + nb, vWM[j], sM);
            #pragma unroll
            for (int j = 0; j < 4; ++j) { aA[j] += TILE_K * 4; vWM[j] += TILE_K * 4; }
        }

        // ---- convert raw fp32 -> bf16 tiles (quantize W on the fly) -------
        {
            const float* rA = (const float*)(smem + bufoff);
            const float* rW = (const float*)(smem + bufoff + RAW_W_OFF);
            const float* rM = (const float*)(smem + bufoff + RAW_M_OFF);
            #pragma unroll
            for (int i = 0; i < 16; ++i)
                lA[cm * TILE_K + ckh + i] = (__bf16)rA[cm * TILE_K + ckh + i];
            #pragma unroll
            for (int i = 0; i < 16; ++i) {
                float w = rW[cm * TILE_K + ckh + i];
                float m = rM[cm * TILE_K + ckh + i];
                // ternary STE: clip(round(w/0.5), -1, 1) (rintf == RNE == jnp.round)
                float q = fminf(1.0f, fmaxf(-1.0f, rintf(w * 2.0f))) * sc * m;
                lB[(ckh + i) * TILE_N + cm] = (__bf16)q;
            }
        }
        __syncthreads();         // bf16 tiles ready

        // ---- 8 WMMAs per wave, B operand reused ---------------------------
        v16bf b;
        #pragma unroll
        for (int j = 0; j < 16; ++j)       // lane%16 == N col; elems span K (table order)
            b[j] = lB[(j + 8 * hi + (j >= 8 ? 8 : 0)) * TILE_N + wave * 16 + ar];

        #pragma unroll
        for (int mt = 0; mt < 8; ++mt) {
            v16bf a;
            const int arow = mt * 16 + ar;
            #pragma unroll
            for (int r = 0; r < 8; ++r) {
                int kb = 2 * r + 8 * (hi + (r >= 4 ? 1 : 0));
                a[2 * r]     = lA[arow * TILE_K + kb];
                a[2 * r + 1] = lA[arow * TILE_K + kb + 1];
            }
            c[mt] = __builtin_amdgcn_wmma_f32_16x16x32_bf16(
                        false, a, false, b, (short)0, c[mt], false, false);
        }

        bufoff ^= RAW_STRIDE;
    }

    // ---- store D: VGPR r holds M=r (lanes 0-15) / M=r+8 (lanes 16-31) ------
    const int n = lane & 15;
    #pragma unroll
    for (int mt = 0; mt < 8; ++mt) {
        #pragma unroll
        for (int r = 0; r < 8; ++r) {
            int mm = mt * 16 + r + 8 * hi;
            if (m0 + mm < cnt) {
                int t = toks[mm];
                out[(size_t)t * N_DIM + nbase + wave * 16 + n] = twp[mm] * c[mt][r];
            }
        }
    }
}

// ---------------------------------------------------------------------------
// Host-side launcher (graph-capture safe: only kernel launches on `stream`).
// Inputs: 0=x 1=weight 2=scale 3=expert_masks 4=router_w
// ---------------------------------------------------------------------------
extern "C" void kernel_launch(void* const* d_in, const int* in_sizes, int n_in,
                              void* d_out, int out_size, void* d_ws, size_t ws_size,
                              hipStream_t stream) {
    const float* x      = (const float*)d_in[0];
    const float* weight = (const float*)d_in[1];
    const float* scale  = (const float*)d_in[2];
    const float* masks  = (const float*)d_in[3];
    const float* rw     = (const float*)d_in[4];
    float* out = (float*)d_out;

    char* ws = (char*)d_ws;
    float* top1w  = (float*)ws;                       // 4096 f32
    int*   top1i  = (int*)(ws + T_TOKENS * 4);        // 4096 i32
    int*   counts = (int*)(ws + T_TOKENS * 8);        // 32
    int*   seg    = counts + N_EXP;                   // 32
    int*   cur    = seg + N_EXP;                      // 32
    int*   tlist  = cur + N_EXP;                      // 4096

    moe_init<<<1, 32, 0, stream>>>(counts);
    moe_router<<<T_TOKENS / 8, 256, 0, stream>>>(x, rw, top1w, top1i, counts);
    moe_scan<<<1, 32, 0, stream>>>(counts, seg, cur);
    moe_scatter<<<T_TOKENS / 256, 256, 0, stream>>>(top1i, cur, tlist);

    dim3 grid(N_DIM / TILE_N, (T_TOKENS + TILE_M - 1) / TILE_M, N_EXP);
    moe_gemm<<<grid, 256, 0, stream>>>(x, weight, scale, masks, top1w,
                                       seg, counts, tlist, out);
}
